// AVRRender_48404281426570
// MI455X (gfx1250) — compile-verified, matrix operand order checked by reading, not run on hardware
//
#include <hip/hip_runtime.h>
#include <hip/hip_bf16.h>
#include <math.h>

typedef __attribute__((ext_vector_type(16))) _Float16 v16h;
typedef __attribute__((ext_vector_type(8)))  float    v8f;

// ---------------- constants ----------------
#define N_AZI 32
#define N_ELE 16
#define RAYS  (N_AZI * N_ELE + 2)   // 514
#define S     64                    // samples per ray
#define HID   64
#define LSEQ  512
#define NFREQ (LSEQ / 2 + 1)        // 257
#define NEAR_D 0.1f
#define FAR_D  8.0f
#define SPEED 343.0f
#define FSR   16000.0f
#define TWO_PI 6.283185307179586f

#define GRAYS 4                               // rays batched per workgroup
#define NBLK  ((RAYS + GRAYS - 1) / GRAYS)    // 129 ray-blocks per batch

// workspace byte offsets (all 32B aligned)
#define WS_DIRS   0                          // RAYS*3 f32            (6168 B)
#define WS_TAU    8192                       // S f32                 (256 B)
#define WS_CT     9216                       // S*LSEQ f32            (131072 B)
#define WS_WSIGPK 140288                     // 32*2*32*16 f16 packed (65536 B)
#define WS_BACC   205824                     // bs*S*LSEQ f32         (bs*131072 B)

__device__ __forceinline__ float normdim(float p, float lo, float hi) {
    return 2.0f * (p - lo) / (hi - lo) - 1.0f;
}
__device__ __forceinline__ float denormdim(float q, float lo, float hi) {
    return (q + 1.0f) * 0.5f * (hi - lo) + lo;
}
__device__ __forceinline__ float dval(int s) {
    return (float)s / (float)(S - 1) * (FAR_D - NEAR_D) + NEAR_D;
}
__device__ __forceinline__ float softplusf(float x) {
    return (x > 20.0f) ? x : __logf(1.0f + __expf(x));
}
__device__ __forceinline__ float hash01(int i) {  // stand-in for jax PRNG jitter
    float v = __sinf((float)(i + 1) * 12.9898f) * 43758.5453f;
    return v - floorf(v);
}
// 16-bit fragment mapping: element e of a lane's v16h holds K index kOf(e)
__device__ __forceinline__ int kOf(int e, int khalf) {
    return ((e < 8) ? e : e + 8) + khalf;     // {0..7}|{16..23}, +8 for hi lanes
}

// ---------------- kernel 0: zero the ray-sum accumulator ----------------
__global__ void avr_zero_kernel(float* p, int n) {
    int i = blockIdx.x * blockDim.x + threadIdx.x;
    if (i < n) p[i] = 0.0f;
}

// ---------------- kernel 1: precompute tables ----------------
__global__ void avr_setup_kernel(const float* __restrict__ W_sig,
                                 float* __restrict__ dirs,
                                 float* __restrict__ tau,
                                 float* __restrict__ CT,
                                 _Float16* __restrict__ wsigPack) {
    const int tid = threadIdx.x;
    // ray directions
    for (int i = tid; i < RAYS; i += blockDim.x) {
        float dx, dy, dz;
        if (i < N_AZI * N_ELE) {
            int ai = i / N_ELE, ej = i % N_ELE;
            float azi = TWO_PI * (float)ai / (float)N_AZI
                      + (TWO_PI / (float)N_AZI) * hash01(ai);
            float ele = acosf(2.0f * ((float)(ej + 1) / 17.0f) - 1.0f);
            float se, ce, sa, ca;
            __sincosf(ele, &se, &ce);
            __sincosf(azi, &sa, &ca);
            dx = ca * se; dy = sa * se; dz = ce;
        } else {
            dx = 0.0f; dy = 0.0f; dz = (i == N_AZI * N_ELE) ? 1.0f : -1.0f;
        }
        dirs[i * 3 + 0] = dx; dirs[i * 3 + 1] = dy; dirs[i * 3 + 2] = dz;
    }
    // per-sample fractional receiver delay
    for (int s = tid; s < S; s += blockDim.x)
        tau[s] = FSR * dval(s) / SPEED;
    // combined tail-mask * gathered path-loss table  CT[s][l]
    for (int k = tid; k < S * LSEQ; k += blockDim.x) {
        int s = k >> 9, l = k & (LSEQ - 1);
        float shift = roundf(FSR * dval(s) / SPEED);
        float tail  = (((float)(LSEQ - 1 - l)) - shift > 0.0f) ? 1.0f : 0.0f;
        int   idx   = (int)shift + l;
        if (idx < 4) idx = 5;                  // prev_part = int(0.1/343*16000) = 4
        float ideal = (float)idx / FSR * SPEED;
        CT[k] = tail / (ideal + 0.001f);
    }
    // W_sig -> fragment-ready fp16 pack: [nt][kstep][lane][e] (32B per lane-frag)
    for (int p = tid; p < 32 * 2 * 32 * 16; p += blockDim.x) {
        int e     = p & 15;
        int lane  = (p >> 4) & 31;
        int kstep = (p >> 9) & 1;
        int nt    = p >> 10;
        int col   = nt * 16 + (lane & 15);
        int khalf = (lane < 16) ? 0 : 8;
        int k     = kOf(e, khalf) + kstep * 32;
        wsigPack[p] = (_Float16)W_sig[k * LSEQ + col];
    }
}

// ---------------- kernel 2: fused multi-ray MLP + WMMA GEMM + render epilogue ----------------
__global__ void __launch_bounds__(128)
avr_ray_kernel(const float* __restrict__ rays_o,
               const float* __restrict__ position_tx,
               const float* __restrict__ W1,
               const float* __restrict__ b1,
               const float* __restrict__ w_attn,
               const float* __restrict__ b_sig,
               const float* __restrict__ dirs,
               const float* __restrict__ CT,
               const _Float16* __restrict__ wsigPack,
               float* __restrict__ Bacc) {
    // packed h: [ray][wave(Mtile)][lane][kstep][e]  -> v16h A fragments
    __shared__ __align__(32) _Float16 hpack[GRAYS][4][32][2][16];   // 32 KB
    __shared__ float feats[GRAYS][S][9];
    __shared__ float attnS[GRAYS][S];
    __shared__ float wgt[GRAYS][S];
    __shared__ int   dly[GRAYS][S];

    const int t       = threadIdx.x;
    const int b       = blockIdx.x / NBLK;
    const int rayBase = (blockIdx.x % NBLK) * GRAYS;
    const int gcnt    = min(GRAYS, RAYS - rayBase);

    const float ox = rays_o[b * 3 + 0], oy = rays_o[b * 3 + 1], oz = rays_o[b * 3 + 2];
    const float ntx = normdim(position_tx[b * 3 + 0], -5.0f, 5.0f);
    const float nty = normdim(position_tx[b * 3 + 1], -5.0f, 5.0f);
    const float ntz = normdim(position_tx[b * 3 + 2], -3.0f, 3.0f);

    // per-(ray, sample) features
    for (int idx = t; idx < GRAYS * S; idx += blockDim.x) {
        int g = idx >> 6, s = idx & (S - 1);
        if (g >= gcnt) continue;
        int r = rayBase + g;
        float dx = dirs[r * 3 + 0], dy = dirs[r * 3 + 1], dz = dirs[r * 3 + 2];
        float d = dval(s);
        feats[g][s][0] = normdim(ox + dx * d, -5.0f, 5.0f);
        feats[g][s][1] = normdim(oy + dy * d, -5.0f, 5.0f);
        feats[g][s][2] = normdim(oz + dz * d, -3.0f, 3.0f);
        feats[g][s][3] = -dx; feats[g][s][4] = -dy; feats[g][s][5] = -dz;
        feats[g][s][6] = ntx; feats[g][s][7] = nty; feats[g][s][8] = ntz;
    }
    __syncthreads();

    // h = relu(feats@W1+b1) packed into A-fragment layout; fused attention dot + tx delay
    for (int idx = t; idx < GRAYS * S; idx += blockDim.x) {
        int g = idx >> 6, s = idx & (S - 1);
        if (g >= gcnt) continue;
        const int wave = s >> 4, m = s & 15;
        float attn = 0.0f;
        #pragma unroll 8
        for (int hd = 0; hd < HID; ++hd) {
            float acc = b1[hd];
            #pragma unroll
            for (int f = 0; f < 9; ++f)
                acc = fmaf(feats[g][s][f], W1[f * HID + hd], acc);
            float hval = fmaxf(acc, 0.0f);
            attn = fmaf(hval, w_attn[hd], attn);
            // inverse fragment map: k -> (lane half, element)
            int kstep = hd >> 5, k = hd & 31;
            int half  = (k >> 3) & 1;              // 0..7->lo, 8..15->hi, 16..23->lo, 24..31->hi
            int e     = (k < 8) ? k : ((k < 16) ? k - 8 : ((k < 24) ? k - 8 : k - 16));
            hpack[g][wave][half * 16 + m][kstep][e] = (_Float16)hval;
        }
        attnS[g][s] = softplusf(attn);
        float ddx = denormdim(ntx - feats[g][s][0], -5.0f, 5.0f);
        float ddy = denormdim(nty - feats[g][s][1], -5.0f, 5.0f);
        float ddz = denormdim(ntz - feats[g][s][2], -3.0f, 3.0f);
        float nrm = sqrtf(ddx * ddx + ddy * ddy + ddz * ddz);
        int dd = (int)roundf(nrm * FSR / SPEED);
        dly[g][s] = min(max(dd, 0), LSEQ - 1);
    }
    __syncthreads();

    // cumulative transmittance -> render weights (tiny per-ray scan)
    if (t < GRAYS && t < gcnt) {
        float T = 1.0f;
        const float dstep = (FAR_D - NEAR_D) / (float)(S - 1);
        for (int s = 0; s < S; ++s) {
            float dist  = (s == S - 1) ? 1e10f : dstep;
            float alpha = 1.0f - __expf(-attnS[t][s] * dist);
            wgt[t][s] = T * alpha;
            T *= (1.0f - alpha + 1e-6f);
        }
    }
    __syncthreads();

    // ---- WMMA: sig = h @ W_sig, B fragments shared across GRAYS rays ----
    const int wave = t >> 5;          // 4 waves, each owns one 16-sample M tile
    const int lane = t & 31;
    const int col0 = (lane & 15);

    float* baccb = Bacc + (size_t)b * (S * LSEQ);

    for (int nt = 0; nt < LSEQ / 16; ++nt) {
        const int col = nt * 16 + col0;
        // fragment-ready B: one 32B vector load per k-step (2x global_load_b128)
        v16h bf0 = *(const v16h*)(wsigPack + (((nt * 2 + 0) * 32 + lane) << 4));
        v16h bf1 = *(const v16h*)(wsigPack + (((nt * 2 + 1) * 32 + lane) << 4));
        const float bs_l = b_sig[col];

        float vacc[8];
        #pragma unroll
        for (int i = 0; i < 8; ++i) vacc[i] = 0.0f;

        for (int g = 0; g < gcnt; ++g) {
            // fragment-ready A from LDS (2x ds_load_b128 each)
            v16h a0 = *(const v16h*)(&hpack[g][wave][lane][0][0]);
            v16h a1 = *(const v16h*)(&hpack[g][wave][lane][1][0]);
            v8f c = {};
            c = __builtin_amdgcn_wmma_f32_16x16x32_f16(false, a0, false, bf0,
                                                       (short)0, c, false, false);
            c = __builtin_amdgcn_wmma_f32_16x16x32_f16(false, a1, false, bf1,
                                                       (short)0, c, false, false);
            #pragma unroll
            for (int i = 0; i < 8; ++i) {
                int s = wave * 16 + i + ((lane < 16) ? 0 : 8);
                float gate = CT[s * LSEQ + col]
                           * ((col >= dly[g][s]) ? 1.0f : 0.0f) * wgt[g][s];
                vacc[i] = fmaf(c[i] + bs_l, gate, vacc[i]);
            }
        }
        #pragma unroll
        for (int i = 0; i < 8; ++i) {
            int s = wave * 16 + i + ((lane < 16) ? 0 : 8);
            atomicAdd(&baccb[s * LSEQ + col], vacc[i]);  // ray-sum, L2 resident
        }
    }
}

// ---------------- kernel 3: fractional-delay DFT + reduction -> [bs, F, 2] ----------------
__global__ void __launch_bounds__(256)
avr_dft_kernel(const float* __restrict__ Bacc,
               const float* __restrict__ tau,
               float* __restrict__ out) {
    __shared__ float sre[256];
    __shared__ float sim[256];
    const int t = threadIdx.x;
    const int b = blockIdx.x / NFREQ;
    const int f = blockIdx.x % NFREQ;
    const float4* B4 = (const float4*)(Bacc + (size_t)b * (S * LSEQ));
    const float w  = -TWO_PI * (float)f / (float)LSEQ;

    float re = 0.0f, im = 0.0f;
    for (int q = t; q < (S * LSEQ) / 4; q += 256) {     // float4 loads, never cross s
        int k = q << 2;
        int s = k >> 9, l = k & (LSEQ - 1);
        float4 v = B4[q];
        float ts = tau[s];
        #pragma unroll
        for (int j = 0; j < 4; ++j) {
            float vj = (j == 0) ? v.x : (j == 1) ? v.y : (j == 2) ? v.z : v.w;
            float sn, cs;
            __sincosf(w * ((float)(l + j) + ts), &sn, &cs);
            re = fmaf(vj, cs, re);
            im = fmaf(vj, sn, im);
        }
    }
    sre[t] = re; sim[t] = im;
    __syncthreads();
    for (int off = 128; off > 0; off >>= 1) {
        if (t < off) { sre[t] += sre[t + off]; sim[t] += sim[t + off]; }
        __syncthreads();
    }
    if (t == 0) {
        out[(b * NFREQ + f) * 2 + 0] = sre[0];
        out[(b * NFREQ + f) * 2 + 1] = sim[0];
    }
}

// ---------------- host-side launcher ----------------
extern "C" void kernel_launch(void* const* d_in, const int* in_sizes, int n_in,
                              void* d_out, int out_size, void* d_ws, size_t ws_size,
                              hipStream_t stream) {
    const float* rays_o      = (const float*)d_in[0];
    const float* position_tx = (const float*)d_in[1];
    const float* W1          = (const float*)d_in[2];
    const float* b1          = (const float*)d_in[3];
    const float* w_attn      = (const float*)d_in[4];
    const float* W_sig       = (const float*)d_in[5];
    const float* b_sig       = (const float*)d_in[6];
    const int bs = in_sizes[0] / 3;

    char* ws = (char*)d_ws;
    float*    dirs     = (float*)(ws + WS_DIRS);
    float*    tau      = (float*)(ws + WS_TAU);
    float*    CT       = (float*)(ws + WS_CT);
    _Float16* wsigPack = (_Float16*)(ws + WS_WSIGPK);
    float*    Bacc     = (float*)(ws + WS_BACC);

    const int nacc = bs * S * LSEQ;
    avr_zero_kernel<<<(nacc + 255) / 256, 256, 0, stream>>>(Bacc, nacc);
    avr_setup_kernel<<<1, 256, 0, stream>>>(W_sig, dirs, tau, CT, wsigPack);
    avr_ray_kernel<<<bs * NBLK, 128, 0, stream>>>(rays_o, position_tx, W1, b1,
                                                  w_attn, b_sig, dirs, CT,
                                                  wsigPack, Bacc);
    avr_dft_kernel<<<bs * NFREQ, 256, 0, stream>>>(Bacc, tau, (float*)d_out);
}
